// SimpleGCN_51445118271859
// MI455X (gfx1250) — matrix operand order repaired and checked by reading, not compiled, not run on hardware
//
#include <hip/hip_runtime.h>

// ---------------------------------------------------------------------------
// GCN on gfx1250 (MI455X): bf16 WMMA GEMMs + L2-resident atomic aggregation.
// D = A(MxK, bf16) * B(KxN, bf16) with f32 accumulation via
// v_wmma_f32_16x16x32_bf16. One 16x16 tile per wave32.
// ---------------------------------------------------------------------------

typedef __attribute__((ext_vector_type(16))) __bf16 v16bf;
typedef __attribute__((ext_vector_type(8)))  __bf16 v8bf;
typedef __attribute__((ext_vector_type(8)))  float  v8f;

#define TPB 256

// ---------------- small elementwise kernels --------------------------------

__global__ void k_fill(float* __restrict__ p, float v, int n) {
    int i = blockIdx.x * blockDim.x + threadIdx.x;
    if (i < n) p[i] = v;
}

__global__ void k_deg_count(float* __restrict__ deg, const int* __restrict__ dst, int e) {
    int i = blockIdx.x * blockDim.x + threadIdx.x;
    if (i < e) atomicAdd(&deg[dst[i]], 1.0f);
}

__global__ void k_rsqrt_inplace(float* __restrict__ p, int n) {
    int i = blockIdx.x * blockDim.x + threadIdx.x;
    if (i < n) p[i] = rsqrtf(p[i]);
}

// W (K x Nout, row-major, f32) -> WT (Nout x K, row-major, bf16)
__global__ void k_weight_t_bf16(const float* __restrict__ W, __bf16* __restrict__ WT,
                                int K, int Nout) {
    int i = blockIdx.x * blockDim.x + threadIdx.x;
    if (i < K * Nout) {
        int k = i / Nout, n = i - k * Nout;
        WT[(size_t)n * K + k] = (__bf16)W[i];
    }
}

__global__ void k_cvt_bf16(const float* __restrict__ s, __bf16* __restrict__ d, int n) {
    int i = blockIdx.x * blockDim.x + threadIdx.x;
    if (i < n) d[i] = (__bf16)s[i];
}

__global__ void k_relu_cvt_bf16(const float* __restrict__ s, __bf16* __restrict__ d, int n) {
    int i = blockIdx.x * blockDim.x + threadIdx.x;
    if (i < n) d[i] = (__bf16)fmaxf(s[i], 0.0f);
}

// agg[i,f] = bias[f] + h[i,f] * dinv[i]^2   (bias + self-loop term, no atomics)
__global__ void k_init_agg(float* __restrict__ agg, const float* __restrict__ h,
                           const float* __restrict__ bias, const float* __restrict__ dinv,
                           int n, int Hdim) {
    int i = blockIdx.x * blockDim.x + threadIdx.x;
    if (i < n * Hdim) {
        int row = i / Hdim, f = i - row * Hdim;
        float di = dinv[row];
        agg[i] = bias[f] + h[i] * di * di;
    }
}

// One block (256 threads) per edge; Hdim == blockDim.x == 256.
// agg[dst] += h[src] * dinv[src]*dinv[dst]  (f32 atomics, resolved in L2)
__global__ void k_scatter_edges(float* __restrict__ agg, const float* __restrict__ h,
                                const float* __restrict__ dinv,
                                const int* __restrict__ src, const int* __restrict__ dst,
                                int Hdim) {
    int e = blockIdx.x;
    int s = src[e], d = dst[e];
    float w = dinv[s] * dinv[d];
    int f = threadIdx.x;
    atomicAdd(&agg[(size_t)d * Hdim + f], h[(size_t)s * Hdim + f] * w);
}

// ---------------- WMMA GEMM -------------------------------------------------
// A: M x K row-major bf16. BT: Nn x K row-major bf16 (i.e. B transposed).
// C: M x Nn f32 (optionally + bias). M%16==0, K%32==0, Nn%16==0 guaranteed.
// 8 waves per block, one 16x16 output tile per wave.

__global__ __launch_bounds__(TPB)
void k_gemm_wmma_bf16(const __bf16* __restrict__ A, const __bf16* __restrict__ BT,
                      const float* __restrict__ bias, float* __restrict__ C,
                      int M, int Nn, int K, int addBias) {
    const int lane = threadIdx.x & 31;
    const int wave = threadIdx.x >> 5;
    const int tilesN = Nn >> 4;
    const int tilesM = M >> 4;
    int tile = blockIdx.x * (TPB / 32) + wave;
    if (tile >= tilesM * tilesN) return;   // wave-uniform: EXEC stays all-ones
    const int tm = tile / tilesN;
    const int tn = tile - tm * tilesN;

    const int half = lane >> 4;   // 0: lanes 0-15, 1: lanes 16-31
    const int l16  = lane & 15;

    // A fragment base (contiguous K order, matching B):
    //   element e of v16bf holds K = k0 + half*16 + e
    const __bf16* Ap = A  + (size_t)(tm * 16 + l16) * K + half * 16;
    // B fragment base (contiguous K order, matching A):
    //   element e of v16bf holds K = k0 + half*16 + e, column = tn*16 + l16
    const __bf16* Bp = BT + (size_t)(tn * 16 + l16) * K + half * 16;

    v8f c = {};
    for (int k0 = 0; k0 < K; k0 += 32) {
        v16bf a = *(const v16bf*)(Ap + k0);        // K = k0+half*16 + 0..15
        v16bf b = *(const v16bf*)(Bp + k0);        // 32B contiguous column chunk
        __builtin_prefetch(Ap + k0 + 32, 0, 1);    // global_prefetch_b8 ahead
        c = __builtin_amdgcn_wmma_f32_16x16x32_bf16(
                /*neg_a=*/false, a, /*neg_b=*/false, b,
                /*c_mod=*/(short)0, c, /*reuse_a=*/false, /*reuse_b=*/false);
    }

    // D layout: VGPR r holds M = tm*16 + half*8 + r, N = tn*16 + l16
    const int n = tn * 16 + l16;
    const float bv = addBias ? bias[n] : 0.0f;
    const size_t rowbase = (size_t)(tm * 16 + half * 8);
#pragma unroll
    for (int r = 0; r < 8; ++r) {
        C[(rowbase + r) * Nn + n] = c[r] + bv;
    }
}

// ---------------- host orchestration ---------------------------------------

static inline int cdiv(int a, int b) { return (a + b - 1) / b; }
static inline size_t alignup(size_t x) { return (x + 255) & ~(size_t)255; }

extern "C" void kernel_launch(void* const* d_in, const int* in_sizes, int n_in,
                              void* d_out, int out_size, void* d_ws, size_t ws_size,
                              hipStream_t stream) {
    const int N = 50000, F = 128, H = 256, Cc = 32;
    const int E = in_sizes[1] / 2;

    const float* x  = (const float*)d_in[0];
    const int*   ei = (const int*)d_in[1];       // [2, E] int32
    const float* W1 = (const float*)d_in[2];     // [128,256]
    const float* b1 = (const float*)d_in[3];
    const float* W2 = (const float*)d_in[4];     // [256,256]
    const float* b2 = (const float*)d_in[5];
    const float* Wc = (const float*)d_in[6];     // [256,32]
    const float* bc = (const float*)d_in[7];
    float* out = (float*)d_out;

    const int* src = ei;
    const int* dst = ei + E;

    // workspace carve
    char* ws = (char*)d_ws;
    float*  dinv = (float*)ws;                 ws += alignup((size_t)N * 4);
    __bf16* W1T  = (__bf16*)ws;                ws += alignup((size_t)F * H * 2);
    __bf16* W2T  = (__bf16*)ws;                ws += alignup((size_t)H * H * 2);
    __bf16* WcT  = (__bf16*)ws;                ws += alignup((size_t)H * Cc * 2);
    __bf16* Abf  = (__bf16*)ws;                ws += alignup((size_t)N * H * 2);  // reused per layer
    float*  hbuf = (float*)ws;                 ws += alignup((size_t)N * H * 4);
    float*  agg  = (float*)ws;                 ws += alignup((size_t)N * H * 4);
    (void)ws_size; (void)n_in; (void)out_size;

    // degrees (self-loop = initial 1.0), then dinv = rsqrt(deg)
    k_fill<<<cdiv(N, TPB), TPB, 0, stream>>>(dinv, 1.0f, N);
    k_deg_count<<<cdiv(E, TPB), TPB, 0, stream>>>(dinv, dst, E);
    k_rsqrt_inplace<<<cdiv(N, TPB), TPB, 0, stream>>>(dinv, N);

    // weights -> transposed bf16
    k_weight_t_bf16<<<cdiv(F * H, TPB), TPB, 0, stream>>>(W1, W1T, F, H);
    k_weight_t_bf16<<<cdiv(H * H, TPB), TPB, 0, stream>>>(W2, W2T, H, H);
    k_weight_t_bf16<<<cdiv(H * Cc, TPB), TPB, 0, stream>>>(Wc, WcT, H, Cc);

    // ---- layer 1: h = x @ W1 ; agg = scatter(h) + b1 ; A2 = bf16(relu(agg))
    k_cvt_bf16<<<cdiv(N * F, TPB), TPB, 0, stream>>>(x, Abf, N * F);
    {
        int tiles = (N / 16) * (H / 16);
        k_gemm_wmma_bf16<<<cdiv(tiles, TPB / 32), TPB, 0, stream>>>(
            Abf, W1T, nullptr, hbuf, N, H, F, 0);
    }
    k_init_agg<<<cdiv(N * H, TPB), TPB, 0, stream>>>(agg, hbuf, b1, dinv, N, H);
    k_scatter_edges<<<E, H, 0, stream>>>(agg, hbuf, dinv, src, dst, H);
    k_relu_cvt_bf16<<<cdiv(N * H, TPB), TPB, 0, stream>>>(agg, Abf, N * H);

    // ---- layer 2: h = A2 @ W2 ; agg = scatter(h) + b2 ; A3 = bf16(relu(agg))
    {
        int tiles = (N / 16) * (H / 16);
        k_gemm_wmma_bf16<<<cdiv(tiles, TPB / 32), TPB, 0, stream>>>(
            Abf, W2T, nullptr, hbuf, N, H, H, 0);
    }
    k_init_agg<<<cdiv(N * H, TPB), TPB, 0, stream>>>(agg, hbuf, b2, dinv, N, H);
    k_scatter_edges<<<E, H, 0, stream>>>(agg, hbuf, dinv, src, dst, H);
    k_relu_cvt_bf16<<<cdiv(N * H, TPB), TPB, 0, stream>>>(agg, Abf, N * H);

    // ---- classifier: out = A3 @ Wc + bc
    {
        int tiles = (N / 16) * (Cc / 16);
        k_gemm_wmma_bf16<<<cdiv(tiles, TPB / 32), TPB, 0, stream>>>(
            Abf, WcT, bc, out, N, Cc, H, 1);
    }
}
